// AttentionBlock_19473381720189
// MI455X (gfx1250) — compile-verified, matrix operand order checked by reading, not run on hardware
//
#include <hip/hip_runtime.h>
#include <math.h>

// ---------- types ----------
typedef unsigned short u16;
typedef __attribute__((ext_vector_type(16))) __bf16  v16bf;
typedef __attribute__((ext_vector_type(8)))  float   v8f;
typedef __attribute__((ext_vector_type(4)))  unsigned int u32x4;
typedef __attribute__((ext_vector_type(8)))  int     i32x8;
typedef __attribute__((ext_vector_type(4)))  int     i32x4;

union Frag {           // one WMMA 16-bit A/B operand: 16 bf16 per lane
    u32x4 u[2];
    v16bf f;
};

// ---------- helpers ----------
__device__ __forceinline__ u16 f2bf(float x) {
    union { float f; unsigned u; } c; c.f = x;
    unsigned u = c.u;
    u += 0x7fffu + ((u >> 16) & 1u);       // round-to-nearest-even
    return (u16)(u >> 16);
}
__device__ __forceinline__ float bf2f(u16 h) {
    union { unsigned u; float f; } c; c.u = ((unsigned)h) << 16;
    return c.f;
}

// ---------- TDM: DMA a [128 x 32]-bf16 tile (row stride K elems) into LDS ----------
// D# per CDNA5 ISA 8.3-8.6: group0 = {count|flags, lds_addr, global_addr[31:0],
// global_addr[56:32]|type=2}; group1 = {data_size=2B, tensor_dim0/1, tile_dim0=32,
// tile_dim1=128, tensor_dim0_stride=K}; groups 2/3 zero (2-D tensor).
// Toolchain here is the 6-arg clang-23 flavor: (u32x4, i32x8, i32x4, i32x4, i32x8, cpol).
__device__ __forceinline__ void tdm_load_tile(const u16* gptr, unsigned ldsOff,
                                              int K, int Mtot) {
    unsigned long long ga = (unsigned long long)(uintptr_t)gptr;   // byte address
    u32x4 g0;
    g0.x = 1u;                                                   // count=1, load, user
    g0.y = ldsOff;                                               // lds_addr (bytes)
    g0.z = (unsigned)ga;                                         // global_addr[31:0]
    g0.w = (unsigned)((ga >> 32) & 0x01ffffffu) | (2u << 30);    // addr[56:32] | type=2
    i32x8 g1;
    g1[0] = 0x10000;                                             // data_size=1 (2 bytes)
    g1[1] = (int)(((unsigned)K & 0xffffu) << 16);                // tensor_dim0 lo16
    g1[2] = (int)((((unsigned)K >> 16) & 0xffffu)
                 | (((unsigned)Mtot & 0xffffu) << 16));          // dim0 hi16 | dim1 lo16
    g1[3] = (int)((((unsigned)Mtot >> 16) & 0xffffu)
                 | (32u << 16));                                 // dim1 hi16 | tile_dim0=32
    g1[4] = 128;                                                 // tile_dim1=128, tile_dim2=0
    g1[5] = K;                                                   // tensor_dim0_stride lo32
    g1[6] = 0;                                                   // stride hi16 | dim1_stride lo
    g1[7] = 0;
    i32x4 z4 = {0, 0, 0, 0};
    i32x8 z8 = {0, 0, 0, 0, 0, 0, 0, 0};
    __builtin_amdgcn_tensor_load_to_lds(g0, g1, z4, z4, z8, 0);
}

// ---------- elementwise convert ----------
__global__ void cvt_f32_bf16(const float* __restrict__ s, u16* __restrict__ d, int n) {
    int i = blockIdx.x * blockDim.x + threadIdx.x;
    if (i < n) d[i] = f2bf(s[i]);
}

// ---------- RoPE in-place on bf16 [rows x (nh*64)] ----------
__global__ void rope_bf16(u16* __restrict__ t, const int* __restrict__ pos_ids,
                          int rows, int nh, int S) {
    int tid = blockIdx.x * blockDim.x + threadIdx.x;
    int total = rows * nh * 32;
    if (tid >= total) return;
    int i   = tid & 31;
    int h   = (tid >> 5) % nh;
    int row = tid / (nh * 32);
    int s   = row % S;
    float p = (float)pos_ids[s];
    float inv_freq = __expf(-logf(10000.0f) * (float)(2 * i) / 64.0f);
    float ang = p * inv_freq;
    float sn, cs;
    __sincosf(ang, &sn, &cs);
    size_t base = (size_t)row * (size_t)(nh * 64) + (size_t)h * 64;
    float x1 = bf2f(t[base + i]);
    float x2 = bf2f(t[base + 32 + i]);
    t[base + i]      = f2bf(x1 * cs - x2 * sn);
    t[base + 32 + i] = f2bf(x1 * sn + x2 * cs);
}

// ---------- GEMM: C[M,N] = A[M,K](bf16) . W[N,K](bf16)^T ----------
// 256 threads = 8 waves; block tile 128x128; wave tile 32x64 (8 WMMA accums).
// K-chunks of 32 staged to LDS by the Tensor Data Mover, double-buffered.
__global__ __launch_bounds__(256)
void gemm_bf16(const u16* __restrict__ A, const u16* __restrict__ W,
               void* __restrict__ Cout, int M, int N, int K, int outBf16) {
    __shared__ __align__(16) u16 ldsA[2][128 * 32];
    __shared__ __align__(16) u16 ldsB[2][128 * 32];

    int wave = threadIdx.x >> 5, lane = threadIdx.x & 31;
    int lm = lane & 15, kh = lane >> 4;
    int wm = wave & 3, wn = wave >> 2;
    int row0 = blockIdx.y * 128;
    int col0 = blockIdx.x * 128;

    v8f c[2][4];
    #pragma unroll
    for (int s2 = 0; s2 < 2; ++s2)
        #pragma unroll
        for (int t = 0; t < 4; ++t) c[s2][t] = v8f{};

    unsigned offA[2] = { (unsigned)(uintptr_t)&ldsA[0][0], (unsigned)(uintptr_t)&ldsA[1][0] };
    unsigned offB[2] = { (unsigned)(uintptr_t)&ldsB[0][0], (unsigned)(uintptr_t)&ldsB[1][0] };

    int nch = K >> 5;
    if (wave == 0) {                       // prologue: stage chunk 0 into buffer 0
        tdm_load_tile(A + (size_t)row0 * K, offA[0], K, M);
        tdm_load_tile(W + (size_t)col0 * K, offB[0], K, N);
    }

    for (int i = 0; i < nch; ++i) {
        int cur = i & 1, nxt = cur ^ 1;
        if (wave == 0) {
            if (i + 1 < nch) {             // prefetch next chunk via TDM, then wait for cur
                tdm_load_tile(A + (size_t)row0 * K + (i + 1) * 32, offA[nxt], K, M);
                tdm_load_tile(W + (size_t)col0 * K + (i + 1) * 32, offB[nxt], K, N);
                __builtin_amdgcn_s_wait_tensorcnt(2);
            } else {
                __builtin_amdgcn_s_wait_tensorcnt(0);
            }
        }
        __syncthreads();                   // chunk i visible to all waves

        const u16* pa = &ldsA[cur][0];
        const u16* pb = &ldsB[cur][0];
        Frag a[2];
        #pragma unroll
        for (int s2 = 0; s2 < 2; ++s2) {
            const u16* ar = pa + (wm * 32 + s2 * 16 + lm) * 32;
            a[s2].u[0] = *(const u32x4*)(ar + kh * 8);
            a[s2].u[1] = *(const u32x4*)(ar + 16 + kh * 8);
        }
        #pragma unroll
        for (int t = 0; t < 4; ++t) {
            Frag bfr;
            const u16* br = pb + (wn * 64 + t * 16 + lm) * 32;
            bfr.u[0] = *(const u32x4*)(br + kh * 8);
            bfr.u[1] = *(const u32x4*)(br + 16 + kh * 8);
            c[0][t] = __builtin_amdgcn_wmma_f32_16x16x32_bf16(false, a[0].f, false, bfr.f,
                                                              (short)0, c[0][t], false, false);
            c[1][t] = __builtin_amdgcn_wmma_f32_16x16x32_bf16(false, a[1].f, false, bfr.f,
                                                              (short)0, c[1][t], false, false);
        }
        __syncthreads();                   // all reads of buf cur done before it is re-staged
    }

    // store: C layout -> VGPR r: lanes0-15 M=r, lanes16-31 M=r+8; N = lane%16
    #pragma unroll
    for (int s2 = 0; s2 < 2; ++s2) {
        #pragma unroll
        for (int t = 0; t < 4; ++t) {
            #pragma unroll
            for (int r = 0; r < 8; ++r) {
                int grow = row0 + wm * 32 + s2 * 16 + r + 8 * kh;
                int gcol = col0 + wn * 64 + t * 16 + lm;
                if (outBf16) ((u16*)Cout)[(size_t)grow * N + gcol]   = f2bf(c[s2][t][r]);
                else         ((float*)Cout)[(size_t)grow * N + gcol] = c[s2][t][r];
            }
        }
    }
}

// ---------- flash attention (causal, GQA 32q/8kv, D=64) ----------
// grid: (S/64, 32, B); block 128 = 4 waves, wave -> 16 q rows.
__global__ __launch_bounds__(128)
void attn_kernel(const u16* __restrict__ qb, const u16* __restrict__ kb,
                 const u16* __restrict__ vb, u16* __restrict__ ob) {
    const int S = 2048;
    int bx = blockIdx.x, h = blockIdx.y, b = blockIdx.z;
    int wave = threadIdx.x >> 5, lane = threadIdx.x & 31;
    int lm = lane & 15, kh = lane >> 4;
    int q0  = bx * 64 + wave * 16;
    int kvh = h >> 2;

    __shared__ __align__(16) u16 vT[64 * 32];       // [dim][kp] transposed V tile
    __shared__ __align__(16) u16 pS[4][16 * 32];    // per-wave P tiles

    const u16* qrow = qb + (size_t)(b * S + q0 + lm) * 2048 + h * 64;
    Frag a0, a1;
    a0.u[0] = *(const u32x4*)(qrow + kh * 8);
    a0.u[1] = *(const u32x4*)(qrow + 16 + kh * 8);
    a1.u[0] = *(const u32x4*)(qrow + 32 + kh * 8);
    a1.u[1] = *(const u32x4*)(qrow + 48 + kh * 8);

    v8f o[4]; float mrun[8], lrun[8];
    #pragma unroll
    for (int g = 0; g < 4; ++g) o[g] = v8f{};
    #pragma unroll
    for (int r = 0; r < 8; ++r) { mrun[r] = -INFINITY; lrun[r] = 0.f; }

    int ntiles = 2 * bx + 2;
    for (int kt = 0; kt < ntiles; ++kt) {
        int kbase = kt * 32;

        __syncthreads();
        {   // cooperative transposed V stage: thread -> row kp, 16 dims
            int kp   = threadIdx.x >> 2;
            int dim0 = (threadIdx.x & 3) * 16;
            const u16* vrow = vb + (size_t)(b * S + kbase + kp) * 512 + kvh * 64 + dim0;
            u32x4 d0 = *(const u32x4*)(vrow);
            u32x4 d1 = *(const u32x4*)(vrow + 8);
            u16 tmp[16];
            *(u32x4*)tmp       = d0;
            *(u32x4*)(tmp + 8) = d1;
            #pragma unroll
            for (int j = 0; j < 16; ++j) vT[(dim0 + j) * 32 + kp] = tmp[j];
        }
        __syncthreads();

        if (kbase > q0 + 15) continue;               // fully masked for this wave

        const u16* k0r = kb + (size_t)(b * S + kbase + lm) * 512 + kvh * 64;
        const u16* k1r = k0r + 16 * 512;
        Frag b00, b01, b10, b11;
        b00.u[0] = *(const u32x4*)(k0r + kh * 8);
        b00.u[1] = *(const u32x4*)(k0r + 16 + kh * 8);
        b01.u[0] = *(const u32x4*)(k0r + 32 + kh * 8);
        b01.u[1] = *(const u32x4*)(k0r + 48 + kh * 8);
        b10.u[0] = *(const u32x4*)(k1r + kh * 8);
        b10.u[1] = *(const u32x4*)(k1r + 16 + kh * 8);
        b11.u[0] = *(const u32x4*)(k1r + 32 + kh * 8);
        b11.u[1] = *(const u32x4*)(k1r + 48 + kh * 8);

        v8f s0 = {}, s1 = {};
        s0 = __builtin_amdgcn_wmma_f32_16x16x32_bf16(false, a0.f, false, b00.f, (short)0, s0, false, false);
        s0 = __builtin_amdgcn_wmma_f32_16x16x32_bf16(false, a1.f, false, b01.f, (short)0, s0, false, false);
        s1 = __builtin_amdgcn_wmma_f32_16x16x32_bf16(false, a0.f, false, b10.f, (short)0, s1, false, false);
        s1 = __builtin_amdgcn_wmma_f32_16x16x32_bf16(false, a1.f, false, b11.f, (short)0, s1, false, false);

        float p0[8], p1[8], alpha[8];
        #pragma unroll
        for (int r = 0; r < 8; ++r) {
            int qpos = q0 + r + 8 * kh;
            float x0 = s0[r] * 0.125f;
            float x1 = s1[r] * 0.125f;
            if (kbase + lm      > qpos) x0 = -INFINITY;
            if (kbase + 16 + lm > qpos) x1 = -INFINITY;
            float t = fmaxf(x0, x1);
            #pragma unroll
            for (int off = 8; off >= 1; off >>= 1) t = fmaxf(t, __shfl_xor(t, off, 32));
            float nm = fmaxf(mrun[r], t);
            alpha[r] = __expf(mrun[r] - nm);
            mrun[r] = nm;
            float e0 = __expf(x0 - nm), e1 = __expf(x1 - nm);
            p0[r] = e0; p1[r] = e1;
            float rs = e0 + e1;
            #pragma unroll
            for (int off = 8; off >= 1; off >>= 1) rs += __shfl_xor(rs, off, 32);
            lrun[r] = lrun[r] * alpha[r] + rs;
        }

        u16* pw = pS[wave];
        #pragma unroll
        for (int r = 0; r < 8; ++r) {
            int mrow = r + 8 * kh;
            pw[mrow * 32 + lm]      = f2bf(p0[r]);
            pw[mrow * 32 + 16 + lm] = f2bf(p1[r]);
        }
        asm volatile("s_wait_dscnt 0" ::: "memory");  // per-wave LDS RAW

        Frag pa;
        pa.u[0] = *(const u32x4*)(pw + lm * 32 + kh * 8);
        pa.u[1] = *(const u32x4*)(pw + lm * 32 + 16 + kh * 8);

        #pragma unroll
        for (int g = 0; g < 4; ++g) {
            #pragma unroll
            for (int r = 0; r < 8; ++r) o[g][r] *= alpha[r];
            Frag bv;
            const u16* vcol = &vT[(g * 16 + lm) * 32];
            bv.u[0] = *(const u32x4*)(vcol + kh * 8);
            bv.u[1] = *(const u32x4*)(vcol + 16 + kh * 8);
            o[g] = __builtin_amdgcn_wmma_f32_16x16x32_bf16(false, pa.f, false, bv.f,
                                                           (short)0, o[g], false, false);
        }
    }

    #pragma unroll
    for (int r = 0; r < 8; ++r) {
        float inv = 1.0f / lrun[r];
        int row = b * S + q0 + r + 8 * kh;
        #pragma unroll
        for (int g = 0; g < 4; ++g) {
            ob[(size_t)row * 2048 + h * 64 + g * 16 + lm] = f2bf(o[g][r] * inv);
        }
    }
}

// ---------- host launcher ----------
extern "C" void kernel_launch(void* const* d_in, const int* in_sizes, int n_in,
                              void* d_out, int out_size, void* d_ws, size_t ws_size,
                              hipStream_t stream) {
    const int B = 2, S = 2048, E = 2048, H = 32, KVH = 8;
    const int BS = B * S;                 // 4096
    const int KVD = KVH * 64;             // 512

    const float* x  = (const float*)d_in[0];
    const int*   pid= (const int*)  d_in[1];
    const float* Wq = (const float*)d_in[2];
    const float* Wk = (const float*)d_in[3];
    const float* Wv = (const float*)d_in[4];
    const float* Wo = (const float*)d_in[5];
    float* out = (float*)d_out;

    char* w = (char*)d_ws;
    u16* xbf  = (u16*)w;                 w += (size_t)BS * E * 2;
    u16* wqbf = (u16*)w;                 w += (size_t)E * E * 2;
    u16* wkbf = (u16*)w;                 w += (size_t)KVD * E * 2;
    u16* wvbf = (u16*)w;                 w += (size_t)KVD * E * 2;
    u16* wobf = (u16*)w;                 w += (size_t)E * E * 2;
    u16* qbf  = (u16*)w;                 w += (size_t)BS * E * 2;
    u16* kbf  = (u16*)w;                 w += (size_t)BS * KVD * 2;
    u16* vbf  = (u16*)w;                 w += (size_t)BS * KVD * 2;
    u16* abf  = (u16*)w;                 w += (size_t)BS * E * 2;
    (void)ws_size; (void)in_sizes; (void)n_in; (void)out_size;

    auto cvt = [&](const float* s, u16* d, int n) {
        cvt_f32_bf16<<<(n + 255) / 256, 256, 0, stream>>>(s, d, n);
    };
    cvt(x,  xbf,  BS * E);
    cvt(Wq, wqbf, E * E);
    cvt(Wk, wkbf, KVD * E);
    cvt(Wv, wvbf, KVD * E);
    cvt(Wo, wobf, E * E);

    // QKV projections (bf16 out), TDM-fed GEMMs
    gemm_bf16<<<dim3(E   / 128, BS / 128), 256, 0, stream>>>(xbf, wqbf, qbf, BS, E,   E, 1);
    gemm_bf16<<<dim3(KVD / 128, BS / 128), 256, 0, stream>>>(xbf, wkbf, kbf, BS, KVD, E, 1);
    gemm_bf16<<<dim3(KVD / 128, BS / 128), 256, 0, stream>>>(xbf, wvbf, vbf, BS, KVD, E, 1);

    // RoPE on q (32 heads) and k (8 heads)
    {
        int nq = BS * H * 32;
        rope_bf16<<<(nq + 255) / 256, 256, 0, stream>>>(qbf, pid, BS, H, S);
        int nk = BS * KVH * 32;
        rope_bf16<<<(nk + 255) / 256, 256, 0, stream>>>(kbf, pid, BS, KVH, S);
    }

    // causal GQA flash attention
    attn_kernel<<<dim3(S / 64, H, B), 128, 0, stream>>>(qbf, kbf, vbf, abf);

    // output projection (f32 out -> d_out)
    gemm_bf16<<<dim3(E / 128, BS / 128), 256, 0, stream>>>(abf, wobf, out, BS, E, E, 0);
}
